// MaskedDecoder_53841710023047
// MI455X (gfx1250) — matrix operand order repaired and checked by reading, not compiled, not run on hardware
//
#include <hip/hip_runtime.h>
#include <hip/hip_bf16.h>

// Problem constants
constexpr int kB  = 8;
constexpr int kNQ = 2048;
constexpr int kNK = 2048;
constexpr int kD  = 512;

typedef __attribute__((ext_vector_type(16))) __bf16 v16bf;
typedef __attribute__((ext_vector_type(8)))  __bf16 v8bf;
typedef __attribute__((ext_vector_type(8)))  float  v8f;

// ---------------------------------------------------------------------------
// Prep: fp32 -> bf16 copies of Q and embs, plus embs^T (so GEMM2 B-fragments
// are contiguous along K).
// ---------------------------------------------------------------------------
__global__ __launch_bounds__(256) void prep_convert(
    const float* __restrict__ Q, const float* __restrict__ E,
    __bf16* __restrict__ Qb, __bf16* __restrict__ Eb, __bf16* __restrict__ EbT) {
  size_t i = (size_t)blockIdx.x * 256 + threadIdx.x;
  size_t total = (size_t)kB * kNK * kD;
  if (i >= total) return;
  float q = Q[i], e = E[i];
  Qb[i] = (__bf16)q;
  Eb[i] = (__bf16)e;
  size_t b   = i / ((size_t)kNK * kD);
  size_t rem = i - b * (size_t)kNK * kD;
  size_t k = rem / kD, d = rem - k * kD;
  EbT[b * (size_t)kD * kNK + d * (size_t)kNK + k] = (__bf16)e;
}

// ---------------------------------------------------------------------------
// Pass A: scores = Q@embs^T + mask, fused softmax over keys -> w [B,NQ,NK]
// One block = one (b, 16-query tile). 8 waves, each wave owns 16 of the 128
// 16x16 column tiles. Scores staged in 128KB dynamic LDS for the softmax.
// ---------------------------------------------------------------------------
__global__ __launch_bounds__(256) void gemm1_softmax(
    const __bf16* __restrict__ Qb, const __bf16* __restrict__ Eb,
    const float* __restrict__ mask, float* __restrict__ w) {
  extern __shared__ float sc[];            // 16 * kNK floats = 128 KB
  int b    = blockIdx.x >> 7;              // kNQ/16 = 128 tiles per batch
  int qt   = blockIdx.x & 127;
  int qbase = qt * 16;
  int tid  = threadIdx.x;
  int wave = tid >> 5, lane = tid & 31;
  int sub  = lane & 15, hi = lane >> 4;

  // A-fragment base: row m = sub; K contiguous: lo-lanes 0..15, hi-lanes 16..31
  const __bf16* arow = Qb + ((size_t)b * kNQ + qbase + sub) * kD + hi * 16;

  for (int jt = wave; jt < kNK / 16; jt += 8) {
    v8f acc = {};
    // B-fragment base: column n = sub (key), K contiguous: lo-lanes 0..15, hi 16..31
    const __bf16* brow = Eb + ((size_t)b * kNK + jt * 16 + sub) * kD + hi * 16;
    for (int kb = 0; kb < kD / 32; ++kb) {
      __builtin_prefetch(brow + (kb + 1) * 32, 0, 3);
      v8bf alo = *(const v8bf*)(arow + kb * 32);
      v8bf ahi = *(const v8bf*)(arow + kb * 32 + 8);
      v16bf afrag;
#pragma unroll
      for (int t = 0; t < 8; ++t) { afrag[t] = alo[t]; afrag[t + 8] = ahi[t]; }
      v16bf bfrag = *(const v16bf*)(brow + kb * 32);
      acc = __builtin_amdgcn_wmma_f32_16x16x32_bf16(
          false, afrag, false, bfrag, (short)0, acc, false, false);
    }
    // C layout: VGPR r -> row m = r + hi*8, col n = sub
#pragma unroll
    for (int r = 0; r < 8; ++r) {
      int m = r + hi * 8;
      int col = jt * 16 + sub;
      float s = acc[r] + mask[((size_t)b * kNQ + qbase + m) * kNK + col];
      sc[m * kNK + col] = s;
    }
  }
  __syncthreads();

  // Row softmax: 16 threads per row (shfl width 16 stays inside half-wave)
  int row = tid >> 4, s16 = tid & 15;
  float* srow = sc + row * kNK;
  float mmax = -3.4e38f;
  for (int c = s16; c < kNK; c += 16) mmax = fmaxf(mmax, srow[c]);
  for (int off = 8; off >= 1; off >>= 1) mmax = fmaxf(mmax, __shfl_xor(mmax, off, 16));
  float ssum = 0.f;
  for (int c = s16; c < kNK; c += 16) {
    float e = __expf(srow[c] - mmax);
    srow[c] = e;
    ssum += e;
  }
  for (int off = 8; off >= 1; off >>= 1) ssum += __shfl_xor(ssum, off, 16);
  float inv = 1.f / ssum;
  float* wrow = w + ((size_t)b * kNQ + qbase + row) * kNK;
  for (int c = s16; c < kNK; c += 16) wrow[c] = srow[c] * inv;
}

// ---------------------------------------------------------------------------
// Pass B: per-column stats over the query axis.
// mx[k] = colmax(w)+1e-4 ; colmax(w/mx) = (mx-1e-4)/mx ; S[k] = sum_q exp(w/mx - M)
// ---------------------------------------------------------------------------
__global__ __launch_bounds__(256) void colstats(
    const float* __restrict__ w, float* __restrict__ mxA, float* __restrict__ SA) {
  int b = blockIdx.x >> 3;                 // kNK/256 = 8 chunks per batch
  int k = ((blockIdx.x & 7) << 8) + threadIdx.x;
  const float* base = w + (size_t)b * kNQ * kNK + k;
  float cmax = -3.4e38f;
  for (int q = 0; q < kNQ; ++q) cmax = fmaxf(cmax, base[(size_t)q * kNK]);
  float mx = cmax + 1e-4f;
  float M  = cmax / mx;
  float inv = 1.f / mx;
  float S = 0.f;
  for (int q = 0; q < kNQ; ++q) S += __expf(base[(size_t)q * kNK] * inv - M);
  mxA[b * kNK + k] = mx;
  SA[b * kNK + k]  = S;
}

// ---------------------------------------------------------------------------
// Pass C: finish column softmax, then per-row min-subtract / max-divide.
// Writes qm fp32 (output #2) and a bf16 copy for GEMM2.
// ---------------------------------------------------------------------------
__global__ __launch_bounds__(256) void rownorm(
    const float* __restrict__ w, const float* __restrict__ mxA,
    const float* __restrict__ SA, float* __restrict__ qm_out,
    __bf16* __restrict__ qmb) {
  __shared__ float red[8];
  int b = blockIdx.x >> 11;                // kNQ = 2048 rows per batch
  int q = blockIdx.x & 2047;
  size_t rowbase = ((size_t)b * kNQ + q) * kNK;
  const float* wrow = w + rowbase;
  const float* mxr = mxA + b * kNK;
  const float* Sr  = SA + b * kNK;
  int tid = threadIdx.x, wv = tid >> 5, ln = tid & 31;

  float t[8];
#pragma unroll
  for (int i = 0; i < 8; ++i) {
    int c = tid + i * 256;
    float mx = mxr[c];
    float M  = (mx - 1e-4f) / mx;
    t[i] = __expf(wrow[c] / mx - M) / Sr[c];
  }
  // block min
  float rmin = t[0];
#pragma unroll
  for (int i = 1; i < 8; ++i) rmin = fminf(rmin, t[i]);
  for (int off = 16; off >= 1; off >>= 1) rmin = fminf(rmin, __shfl_xor(rmin, off, 32));
  if (ln == 0) red[wv] = rmin;
  __syncthreads();
  rmin = red[0];
#pragma unroll
  for (int i = 1; i < 8; ++i) rmin = fminf(rmin, red[i]);
  __syncthreads();
  // block max of (t - rmin)
  float rmax = t[0] - rmin;
#pragma unroll
  for (int i = 1; i < 8; ++i) rmax = fmaxf(rmax, t[i] - rmin);
  for (int off = 16; off >= 1; off >>= 1) rmax = fmaxf(rmax, __shfl_xor(rmax, off, 32));
  if (ln == 0) red[wv] = rmax;
  __syncthreads();
  rmax = red[0];
#pragma unroll
  for (int i = 1; i < 8; ++i) rmax = fmaxf(rmax, red[i]);
  float invmax = 1.f / rmax;
#pragma unroll
  for (int i = 0; i < 8; ++i) {
    int c = tid + i * 256;
    float qv = (t[i] - rmin) * invmax;
    qm_out[rowbase + c] = qv;
    qmb[rowbase + c]    = (__bf16)qv;
  }
}

// ---------------------------------------------------------------------------
// Pass D: Qr = Q + qm@embs, fused LayerNorm over D. One block per 16-query
// tile; 8 waves cover the 32 output column tiles; 32KB LDS for the LN pass.
// ---------------------------------------------------------------------------
__global__ __launch_bounds__(256) void gemm2_ln(
    const __bf16* __restrict__ qmb, const __bf16* __restrict__ EbT,
    const float* __restrict__ Q, const float* __restrict__ gamma,
    const float* __restrict__ beta, float* __restrict__ Qn) {
  __shared__ float qr[16 * kD];            // 32 KB
  int b    = blockIdx.x >> 7;
  int qt   = blockIdx.x & 127;
  int qbase = qt * 16;
  int tid  = threadIdx.x;
  int wave = tid >> 5, lane = tid & 31;
  int sub  = lane & 15, hi = lane >> 4;

  const __bf16* arow = qmb + ((size_t)b * kNQ + qbase + sub) * kNK + hi * 16;
  for (int dt = wave; dt < kD / 16; dt += 8) {
    v8f acc = {};
    // B element (k, n=dcol) = embs[b,k,dcol] = EbT[b][dcol][k]: contiguous K
    const __bf16* brow = EbT + ((size_t)b * kD + dt * 16 + sub) * kNK + hi * 16;
    for (int kb = 0; kb < kNK / 32; ++kb) {
      __builtin_prefetch(brow + (kb + 1) * 32, 0, 3);
      v8bf alo = *(const v8bf*)(arow + kb * 32);
      v8bf ahi = *(const v8bf*)(arow + kb * 32 + 8);
      v16bf afrag;
#pragma unroll
      for (int t = 0; t < 8; ++t) { afrag[t] = alo[t]; afrag[t + 8] = ahi[t]; }
      v16bf bfrag = *(const v16bf*)(brow + kb * 32);
      acc = __builtin_amdgcn_wmma_f32_16x16x32_bf16(
          false, afrag, false, bfrag, (short)0, acc, false, false);
    }
#pragma unroll
    for (int r = 0; r < 8; ++r) {
      int m = r + hi * 8;
      int dcol = dt * 16 + sub;
      qr[m * kD + dcol] = acc[r] + Q[((size_t)b * kNQ + qbase + m) * kD + dcol];
    }
  }
  __syncthreads();

  // LayerNorm: 16 threads per row, 32 elements each
  int row = tid >> 4, s16 = tid & 15;
  float* rrow = qr + row * kD;
  float sum = 0.f, sq = 0.f;
  for (int c = s16; c < kD; c += 16) { float x = rrow[c]; sum += x; sq += x * x; }
  for (int off = 8; off >= 1; off >>= 1) {
    sum += __shfl_xor(sum, off, 16);
    sq  += __shfl_xor(sq,  off, 16);
  }
  float mean = sum * (1.f / kD);
  float var  = sq * (1.f / kD) - mean * mean;
  float rstd = rsqrtf(var + 1e-3f);
  float* orow = Qn + ((size_t)b * kNQ + qbase + row) * kD;
  for (int c = s16; c < kD; c += 16)
    orow[c] = (rrow[c] - mean) * rstd * gamma[c] + beta[c];
}

// ---------------------------------------------------------------------------
extern "C" void kernel_launch(void* const* d_in, const int* in_sizes, int n_in,
                              void* d_out, int out_size, void* d_ws, size_t ws_size,
                              hipStream_t stream) {
  const float* Q     = (const float*)d_in[0];
  const float* E     = (const float*)d_in[1];
  const float* mask  = (const float*)d_in[2];
  const float* gamma = (const float*)d_in[3];
  const float* beta  = (const float*)d_in[4];

  float* out    = (float*)d_out;
  float* Qn     = out;                                   // [B,NQ,D]
  float* qm_out = out + (size_t)kB * kNQ * kD;           // [B,NQ,NK]

  char* ws = (char*)d_ws;
  size_t o = 0;
  auto carve = [&](size_t bytes) {
    char* p = ws + o;
    o += (bytes + 255) & ~(size_t)255;
    return p;
  };
  __bf16* Qb  = (__bf16*)carve((size_t)kB * kNQ * kD * 2);   // 16 MB
  __bf16* Eb  = (__bf16*)carve((size_t)kB * kNK * kD * 2);   // 16 MB
  __bf16* EbT = (__bf16*)carve((size_t)kB * kD * kNK * 2);   // 16 MB
  float*  w   = (float*)carve((size_t)kB * kNQ * kNK * 4);   // 128 MB
  float*  mxA = (float*)carve((size_t)kB * kNK * 4);
  float*  SA  = (float*)carve((size_t)kB * kNK * 4);
  __bf16* qmb = (__bf16*)carve((size_t)kB * kNQ * kNK * 2);  // 64 MB

  {
    size_t total = (size_t)kB * kNK * kD;
    int blocks = (int)((total + 255) / 256);
    prep_convert<<<blocks, 256, 0, stream>>>(Q, E, Qb, Eb, EbT);
  }
  gemm1_softmax<<<kB * (kNQ / 16), 256, 16 * kNK * sizeof(float), stream>>>(Qb, Eb, mask, w);
  colstats<<<kB * (kNK / 256), 256, 0, stream>>>(w, mxA, SA);
  rownorm<<<kB * kNQ, 256, 0, stream>>>(w, mxA, SA, qm_out, qmb);
  gemm2_ln<<<kB * (kNQ / 16), 256, 0, stream>>>(qmb, EbT, Q, gamma, beta, Qn);
}
